// Codebook_20066087206971
// MI455X (gfx1250) — compile-verified, matrix operand order checked by reading, not run on hardware
//
#include <hip/hip_runtime.h>
#include <hip/hip_bf16.h>

// VQ-VAE codebook quantization for gfx1250 (MI455X).
// Fused distance-GEMM (f16 WMMA, f32 accum) + argmin + gather + loss + histogram.
// Codebook streamed through LDS via double-buffered Tensor Data Mover (TDM).

typedef __attribute__((ext_vector_type(16))) _Float16 v16h;
typedef __attribute__((ext_vector_type(8)))  _Float16 v8h;
typedef __attribute__((ext_vector_type(2)))  _Float16 v2h;
typedef __attribute__((ext_vector_type(8)))  float    v8f;

#define NROWS     65536      // B*H*W = 64*32*32
#define DIMV      64         // code dimension
#define KCODES    1024       // codebook size
#define OUT_ELEMS 4194304    // 64*64*32*32
#define LOSS_OFF  4194304
#define IDX_OFF   4194305
#define PERP_OFF  4259841    // 4194305 + 65536

#define CHUNK  128           // codebook rows resident per LDS buffer
#define NCHUNK (KCODES / CHUNK)
#define ROWH   72            // padded LDS row stride in halfs (bank-conflict avoidance)

union ABFrag { v16h v; v8h h[2]; };

#if __has_builtin(__builtin_amdgcn_tensor_load_to_lds)
#define HAS_TDM 1
typedef __attribute__((ext_vector_type(4))) unsigned int v4u;
typedef __attribute__((ext_vector_type(8))) int          v8i;
typedef __attribute__((ext_vector_type(4))) int          v4i;

// Issue one TDM 2D tile load: CHUNK rows x 128B from cb16 into padded LDS.
// data_size = 8B units; row = 16 units; LDS pad: 4 dwords after every 32 dwords
// (reproduces the ROWH=72-half padded layout). Wave-level op, EXEC ignored.
// 6-arg builtin form (clang-23 / therock-10.0): groups 0..3 + spare + cpol.
__device__ __forceinline__ void tdm_load_chunk(const _Float16* gsrc, void* ldst) {
    const unsigned long long ga = (unsigned long long)(uintptr_t)gsrc;
    const unsigned int       la = (unsigned int)(uintptr_t)ldst;   // LDS byte offset
    v4u g0;
    g0[0] = 1u;                                              // count=1, user mode
    g0[1] = la;                                              // lds_addr
    g0[2] = (unsigned int)(ga & 0xFFFFFFFFu);                // global_addr[31:0]
    g0[3] = (unsigned int)((ga >> 32) & 0x01FFFFFFu)         // global_addr[56:32]
          | 0x80000000u;                                     // type=2 ("image")
    v8i g1;
    g1[0] = (int)((3u << 16) | (1u << 20) | (4u << 22) | (3u << 25));
            // wg_mask=0 | data_size=8B | pad_enable | pad_interval=32dw | pad_amount=4dw
    g1[1] = (int)(16u << 16);      // tensor_dim0[15:0] = 16 (8B units)
    g1[2] = (int)(1024u << 16);    // tensor_dim0 hi = 0 ; tensor_dim1[15:0] = 1024
    g1[3] = (int)(16u << 16);      // tensor_dim1 hi = 0 ; tile_dim0 = 16
    g1[4] = (int)CHUNK;            // tile_dim1 = CHUNK rows ; tile_dim2 = 0
    g1[5] = 16;                    // tensor_dim0_stride = 16 (rows contiguous)
    g1[6] = 0;                     // stride0 hi ; tensor_dim1_stride lo = 0 (2D)
    g1[7] = 0;
    const v4i z4 = {0, 0, 0, 0};               // groups 2/3 unused (2D tensor)
    const v8i z8 = {0, 0, 0, 0, 0, 0, 0, 0};
    __builtin_amdgcn_tensor_load_to_lds(g0, g1, z4, z4, z8, 0);
}
#else
#define HAS_TDM 0
#endif

// ---------------------------------------------------------------------------
// Kernel 1: codebook -> f16 copy + per-code squared norms; zero accumulators.
// ---------------------------------------------------------------------------
__global__ __launch_bounds__(256) void vq_prep(const float* __restrict__ cb,
                                               _Float16* __restrict__ cb16,
                                               float* __restrict__ cbnorm,
                                               float* __restrict__ counts,
                                               float* __restrict__ lossacc) {
    int t = blockIdx.x * 256 + threadIdx.x;
    if (t == 0) *lossacc = 0.0f;
    if (t < KCODES) {
        counts[t] = 0.0f;
        const float* r = cb + (size_t)t * DIMV;
        _Float16*    o = cb16 + (size_t)t * DIMV;
        float s = 0.0f;
        #pragma unroll
        for (int c = 0; c < DIMV; c += 2) {
            float a = r[c], b = r[c + 1];
            s += a * a + b * b;
            v2h p = { (_Float16)a, (_Float16)b };
            *(v2h*)(o + c) = p;
        }
        cbnorm[t] = s;
    }
}

// ---------------------------------------------------------------------------
// Kernel 2: [B,C,H,W] f32 -> [N,64] f16, LDS-tiled transpose (64 rows / block).
// ---------------------------------------------------------------------------
__global__ __launch_bounds__(256) void vq_transpose(const float* __restrict__ in,
                                                    _Float16* __restrict__ x16) {
    __shared__ float lds[64 * 65];
    const int tid = threadIdx.x;
    const int n0  = blockIdx.x * 64;            // 64 rows, always within one b
    const int b   = n0 >> 10;
    const int hw0 = n0 & 1023;
    const float* base = in + (size_t)b * 65536 + hw0;
    #pragma unroll
    for (int k = 0; k < 16; ++k) {              // coalesced reads along W
        int m = k * 256 + tid;
        int c = m >> 6, j = m & 63;
        lds[j * 65 + c] = base[(size_t)c << 10 | (unsigned)j];
    }
    __syncthreads();
    #pragma unroll
    for (int k = 0; k < 8; ++k) {               // coalesced f16-pair writes along C
        int m  = k * 256 + tid;
        int j  = m >> 5, c2 = m & 31;
        v2h p = { (_Float16)lds[j * 65 + 2 * c2],
                  (_Float16)lds[j * 65 + 2 * c2 + 1] };
        *(v2h*)(x16 + (size_t)(n0 + j) * DIMV + 2 * c2) = p;
    }
}

// ---------------------------------------------------------------------------
// Kernel 3: fused distance GEMM (WMMA f16) + argmin + gather + loss + histogram.
// 8 waves/block, 32 rows/wave (two M tiles share each B fragment -> 4 WMMAs per
// 4 ds_load_b128). Codebook double-buffered in LDS via TDM (s_wait_tensorcnt).
// ---------------------------------------------------------------------------
__global__ __launch_bounds__(256) void vq_main(const _Float16* __restrict__ x16,
                                               const _Float16* __restrict__ cb16,
                                               const float* __restrict__ cbnorm,
                                               const float* __restrict__ cb,
                                               const float* __restrict__ in,
                                               float* __restrict__ out,
                                               float* __restrict__ idxout,
                                               float* __restrict__ counts,
                                               float* __restrict__ lossacc) {
    __shared__ __align__(16) _Float16 ldsb[2][CHUNK * ROWH];   // 2 x 18 KB
    const int tid  = threadIdx.x;
    const int wave = tid >> 5;
    const int lane = tid & 31;
    const int hs   = (lane >> 4) & 1;                          // lane half select
    const int r0   = blockIdx.x * 256 + wave * 32;

    // ISA 16-bit A layout: low lanes hold K {0-7,16-23}, high lanes K {8-15,24-31}
    const int kb0 = hs ? 8  : 0;
    const int kb1 = hs ? 24 : 16;

    const _Float16* xr0 = x16 + (size_t)(r0 + (lane & 15)) * DIMV;        // M tile 0
    const _Float16* xr1 = x16 + (size_t)(r0 + 16 + (lane & 15)) * DIMV;   // M tile 1
    ABFrag a0, a1, a2, a3;
    a0.h[0] = *(const v8h*)(xr0 + kb0);      a0.h[1] = *(const v8h*)(xr0 + kb1);
    a1.h[0] = *(const v8h*)(xr0 + 32 + kb0); a1.h[1] = *(const v8h*)(xr0 + 32 + kb1);
    a2.h[0] = *(const v8h*)(xr1 + kb0);      a2.h[1] = *(const v8h*)(xr1 + kb1);
    a3.h[0] = *(const v8h*)(xr1 + 32 + kb0); a3.h[1] = *(const v8h*)(xr1 + 32 + kb1);

    float best[16];
    int   bidx[16];
    #pragma unroll
    for (int j = 0; j < 16; ++j) { best[j] = 3.0e38f; bidx[j] = 0; }

#if HAS_TDM
    if (wave == 0) tdm_load_chunk(cb16, &ldsb[0][0]);          // prime buffer 0
#endif

    for (int chunk = 0; chunk < NCHUNK; ++chunk) {
        const int cur = chunk & 1;
#if HAS_TDM
        if (wave == 0) __builtin_amdgcn_s_wait_tensorcnt(0);   // chunk's DMA done
        __syncthreads();                                       // ...visible to all
        if ((chunk + 1 < NCHUNK) && wave == 0)                 // prefetch next chunk
            tdm_load_chunk(cb16 + (size_t)(chunk + 1) * CHUNK * DIMV,
                           &ldsb[(chunk + 1) & 1][0]);
#else
        __syncthreads();
        {   // cooperative fallback: 256 threads copy CHUNK rows (64 B each)
            const int r = tid >> 1, h = tid & 1;
            const uint4* src = (const uint4*)(cb16 + (size_t)(chunk * CHUNK + r) * DIMV
                                                   + h * 32);
            uint4* dst = (uint4*)(&ldsb[cur][0] + r * ROWH + h * 32);
            #pragma unroll
            for (int j = 0; j < 4; ++j) dst[j] = src[j];
        }
        __syncthreads();
#endif
        #pragma unroll 2
        for (int nt = 0; nt < CHUNK / 16; ++nt) {
            const int nloc = nt * 16 + (lane & 15);
            const _Float16* br = &ldsb[cur][0] + nloc * ROWH;
            ABFrag b0, b1;
            b0.h[0] = *(const v8h*)(br + kb0);
            b0.h[1] = *(const v8h*)(br + kb1);
            b1.h[0] = *(const v8h*)(br + 32 + kb0);
            b1.h[1] = *(const v8h*)(br + 32 + kb1);

            v8f acc0 = {}, acc1 = {};
            acc0 = __builtin_amdgcn_wmma_f32_16x16x32_f16(false, a0.v, false, b0.v,
                                                          (short)0, acc0, false, false);
            acc1 = __builtin_amdgcn_wmma_f32_16x16x32_f16(false, a2.v, false, b0.v,
                                                          (short)0, acc1, false, false);
            acc0 = __builtin_amdgcn_wmma_f32_16x16x32_f16(false, a1.v, false, b1.v,
                                                          (short)0, acc0, false, false);
            acc1 = __builtin_amdgcn_wmma_f32_16x16x32_f16(false, a3.v, false, b1.v,
                                                          (short)0, acc1, false, false);

            const int   ncode = chunk * CHUNK + nloc;
            const float nrm   = cbnorm[ncode];
            #pragma unroll
            for (int j = 0; j < 8; ++j) {
                float d0 = nrm - 2.0f * acc0[j];        // ||x||^2 dropped (row-const)
                float d1 = nrm - 2.0f * acc1[j];
                if (d0 < best[j])     { best[j]     = d0; bidx[j]     = ncode; }
                if (d1 < best[8 + j]) { best[8 + j] = d1; bidx[8 + j] = ncode; }
            }
        }
    }

    // Cross-lane argmin inside each 16-lane half (C layout: low half rows j, high j+8)
    #pragma unroll
    for (int j = 0; j < 16; ++j) {
        #pragma unroll
        for (int off = 1; off < 16; off <<= 1) {
            float od = __shfl_xor(best[j], off, 32);
            int   oi = __shfl_xor(bidx[j], off, 32);
            if (od < best[j] || (od == best[j] && oi < bidx[j])) {
                best[j] = od; bidx[j] = oi;
            }
        }
    }

    // Epilogue: 32 lanes <-> 32 rows; each lane writes one full 64-channel row.
    const int jrow = lane & 7;
    const int tsel = (lane >> 3) & 1;
    const int sel  = tsel * 8 + jrow;
    int code = bidx[0];
    #pragma unroll
    for (int j = 1; j < 16; ++j) if (sel == j) code = bidx[j];   // cndmask chain

    const int row = r0 + tsel * 16 + hs * 8 + jrow;
    const int b   = row >> 10;
    const int hw  = row & 1023;
    const float* e   = cb + (size_t)code * DIMV;
    const float* xin = in  + (size_t)b * 65536 + hw;
    float*       op  = out + (size_t)b * 65536 + hw;
    float ss = 0.0f;
    #pragma unroll 8
    for (int c = 0; c < DIMV; ++c) {
        float q   = e[c];
        float xv  = xin[(size_t)c << 10];
        float dlt = q - xv;
        ss += dlt * dlt;
        op[(size_t)c << 10] = q;                 // q_st == quantized numerically
    }
    __hip_atomic_fetch_add(lossacc, ss, __ATOMIC_RELAXED, __HIP_MEMORY_SCOPE_AGENT);
    idxout[row] = (float)code;
    __hip_atomic_fetch_add(counts + code, 1.0f, __ATOMIC_RELAXED, __HIP_MEMORY_SCOPE_AGENT);
}

// ---------------------------------------------------------------------------
// Kernel 4: loss + perplexity finalize.
// ---------------------------------------------------------------------------
__global__ __launch_bounds__(1024) void vq_finalize(const float* __restrict__ counts,
                                                    const float* __restrict__ lossacc,
                                                    float* __restrict__ dout) {
    __shared__ float red[1024];
    const int t = threadIdx.x;
    float p = counts[t] * (1.0f / 65536.0f);
    red[t] = p * __logf(p + 1e-10f);
    __syncthreads();
    for (int s = 512; s > 0; s >>= 1) {
        if (t < s) red[t] += red[t + s];
        __syncthreads();
    }
    if (t == 0) {
        dout[PERP_OFF] = __expf(-red[0]);
        // loss = q_latent + 0.25*e_latent = 1.25 * mean((q - x)^2) over 4194304 elems
        dout[LOSS_OFF] = 1.25f * (*lossacc) * (1.0f / 4194304.0f);
    }
}

// ---------------------------------------------------------------------------
extern "C" void kernel_launch(void* const* d_in, const int* in_sizes, int n_in,
                              void* d_out, int out_size, void* d_ws, size_t ws_size,
                              hipStream_t stream) {
    const float* in = (const float*)d_in[0];   // [64,64,32,32] f32
    const float* cb = (const float*)d_in[1];   // [1024,64] f32
    float* out = (float*)d_out;

    char* ws = (char*)d_ws;
    _Float16* x16    = (_Float16*)ws;                                     // 8 MB
    _Float16* cb16   = (_Float16*)(ws + (size_t)NROWS * DIMV * 2);        // 128 KB
    float*    cbnorm = (float*)(ws + (size_t)NROWS * DIMV * 2
                                   + (size_t)KCODES * DIMV * 2);          // 4 KB
    float*    counts  = cbnorm + KCODES;                                  // 4 KB
    float*    lossacc = counts + KCODES;                                  // 4 B

    vq_prep     <<<KCODES / 256, 256, 0, stream>>>(cb, cb16, cbnorm, counts, lossacc);
    vq_transpose<<<NROWS / 64,   256, 0, stream>>>(in, x16);
    vq_main     <<<NROWS / 256,  256, 0, stream>>>(x16, cb16, cbnorm, cb, in,
                                                   out, out + IDX_OFF, counts, lossacc);
    vq_finalize <<<1, 1024, 0, stream>>>(counts, lossacc, out);
}
